// VanillmModel_42520176231010
// MI455X (gfx1250) — matrix operand order repaired and checked by reading, not compiled, not run on hardware
//
#include <hip/hip_runtime.h>
#include <hip/hip_bf16.h>
#include <math.h>

typedef __attribute__((ext_vector_type(16))) _Float16 v16h;
typedef __attribute__((ext_vector_type(8)))  _Float16 h8;
typedef __attribute__((ext_vector_type(8)))  float    v8f;

#define DEVINL __device__ __forceinline__

static DEVINL v16h join16(h8 lo, h8 hi) {
    return __builtin_shufflevector(lo, hi, 0,1,2,3,4,5,6,7,8,9,10,11,12,13,14,15);
}

static DEVINL v8f wmma_f16(v16h a, v16h b, v8f c) {
    return __builtin_amdgcn_wmma_f32_16x16x32_f16(false, a, false, b, (short)0, c, false, false);
}

// ---------------- constants ----------------
#define TT   2048
#define BB   4
#define CC   1024
#define HH   16
#define HD   64
#define MM   (BB*TT)          // 8192 rows
#define FF   (4*CC)           // 4096
#define VV   32000
#define EPS  1e-5f
#define NEGBIG (-3.0e38f)

// epilogue flags
#define EPI_BIAS 1
#define EPI_RELU 2
#define EPI_RES  4
#define EPI_O32  8
#define EPI_O16  16

// ---------------- weight packing: f32 [K,N] -> f16 [N,K] ----------------
__global__ __launch_bounds__(256) void pack_t_kernel(const float* __restrict__ src,
                                                     _Float16* __restrict__ dst,
                                                     int N, int K) {
    int j = blockIdx.x * 256 + threadIdx.x;
    if (j >= N * K) return;
    int n = j / K, k = j - n * K;
    dst[j] = (_Float16)src[(size_t)k * N + n];
}

// pack Wq/Wk/Wv [H,C,HD] into f16 [3*C rows (n), C cols (k)]  (n = h*64+d)
__global__ __launch_bounds__(256) void pack_qkv_kernel(const float* __restrict__ Wq,
                                                       const float* __restrict__ Wk,
                                                       const float* __restrict__ Wv,
                                                       _Float16* __restrict__ dst) {
    int j = blockIdx.x * 256 + threadIdx.x;   // j = n*CC + c, n in [0,3*CC)
    if (j >= 3 * CC * CC) return;
    int n = j / CC, c = j - n * CC;
    const float* W = (n < CC) ? Wq : (n < 2 * CC) ? Wk : Wv;
    int nl = n & (CC - 1);
    int h = nl >> 6, d = nl & 63;
    dst[j] = (_Float16)W[((size_t)h * CC + c) * HD + d];
}

// ---------------- block reduction ----------------
static DEVINL float block_sum(float v, float* sred) {
    int tid = threadIdx.x;
    sred[tid] = v; __syncthreads();
    for (int s = 128; s > 0; s >>= 1) { if (tid < s) sred[tid] += sred[tid + s]; __syncthreads(); }
    float r = sred[0]; __syncthreads();
    return r;
}

// ---------------- embed + LN1 (one block per row) ----------------
__global__ __launch_bounds__(256) void embed_ln_kernel(const int* __restrict__ idx,
                                                       const float* __restrict__ tok,
                                                       const float* __restrict__ pos,
                                                       const float* __restrict__ g,
                                                       const float* __restrict__ bta,
                                                       float* __restrict__ x,
                                                       _Float16* __restrict__ xn) {
    __shared__ float sred[256];
    int row = blockIdx.x;
    int t = row & (TT - 1);
    int tid = threadIdx.x;
    int tokid = idx[row];
    float4 tv = ((const float4*)(tok + (size_t)tokid * CC))[tid];
    float4 pv = ((const float4*)(pos + (size_t)t * CC))[tid];
    float v0 = tv.x + pv.x, v1 = tv.y + pv.y, v2 = tv.z + pv.z, v3 = tv.w + pv.w;
    float mu = block_sum(v0 + v1 + v2 + v3, sred) * (1.0f / CC);
    float d0 = v0 - mu, d1 = v1 - mu, d2 = v2 - mu, d3 = v3 - mu;
    float var = block_sum(d0*d0 + d1*d1 + d2*d2 + d3*d3, sred) * (1.0f / CC);
    float rstd = rsqrtf(var + EPS);
    float4 g4 = ((const float4*)g)[tid];
    float4 b4 = ((const float4*)bta)[tid];
    float4* xo = (float4*)(x + (size_t)row * CC);
    xo[tid] = make_float4(v0, v1, v2, v3);
    _Float16* xr = xn + (size_t)row * CC + tid * 4;
    xr[0] = (_Float16)(d0 * rstd * g4.x + b4.x);
    xr[1] = (_Float16)(d1 * rstd * g4.y + b4.y);
    xr[2] = (_Float16)(d2 * rstd * g4.z + b4.z);
    xr[3] = (_Float16)(d3 * rstd * g4.w + b4.w);
}

// ---------------- LN (f32 in -> f16 out) ----------------
__global__ __launch_bounds__(256) void ln_kernel(const float* __restrict__ x,
                                                 const float* __restrict__ g,
                                                 const float* __restrict__ bta,
                                                 _Float16* __restrict__ xn) {
    __shared__ float sred[256];
    int row = blockIdx.x;
    int tid = threadIdx.x;
    float4 v = ((const float4*)(x + (size_t)row * CC))[tid];
    float mu = block_sum(v.x + v.y + v.z + v.w, sred) * (1.0f / CC);
    float d0 = v.x - mu, d1 = v.y - mu, d2 = v.z - mu, d3 = v.w - mu;
    float var = block_sum(d0*d0 + d1*d1 + d2*d2 + d3*d3, sred) * (1.0f / CC);
    float rstd = rsqrtf(var + EPS);
    float4 g4 = ((const float4*)g)[tid];
    float4 b4 = ((const float4*)bta)[tid];
    _Float16* xr = xn + (size_t)row * CC + tid * 4;
    xr[0] = (_Float16)(d0 * rstd * g4.x + b4.x);
    xr[1] = (_Float16)(d1 * rstd * g4.y + b4.y);
    xr[2] = (_Float16)(d2 * rstd * g4.z + b4.z);
    xr[3] = (_Float16)(d3 * rstd * g4.w + b4.w);
}

// ============ shared GEMM main loop (async-LDS staged B, 4-stage pipeline) ============
// Block = 4 waves; wave computes 32(M) x 64(N); block = 128(M) x 64(N).
// B tile (64n x 32k = 4KB) loaded once per block per K-step via
// GLOBAL_LOAD_ASYNC_TO_LDS_B128; 4 LDS buffers, 3 tiles in flight.
// ASYNCcnt counts instructions (2 per tile per wave) and async loads complete
// in order, so steady-state wait is asynccnt<=4 (oldest tile done).
struct GemmIdx { int m0, n0, hi, ln; };

static DEVINL GemmIdx gemm_mainloop(const _Float16* __restrict__ A,
                                    const _Float16* __restrict__ Bt,
                                    int K, v8f acc[2][4], _Float16* btile) {
    int tid  = threadIdx.x;
    int lane = tid & 31, widx = tid >> 5;
    int hi = lane >> 4, ln = lane & 15;
    int m0 = blockIdx.y * 128 + widx * 32;
    int n0 = blockIdx.x * 64;
    const _Float16* gB = Bt + (size_t)n0 * K;

    // 64 rows x 64B tile = 256 16B-chunks; 128 threads x 2 chunks each
    int c1t = tid + 128;
    int r0 = tid >> 2,  s0c = tid & 3;
    int r1 = c1t >> 2,  s1c = c1t & 3;
    unsigned vbase0 = (unsigned)(r0 * K * 2 + s0c * 16);
    unsigned vbase1 = (unsigned)(r1 * K * 2 + s1c * 16);
    unsigned lbase  = (unsigned)(size_t)btile;
    unsigned l0 = lbase + (unsigned)(r0 * 64 + s0c * 16);
    unsigned l1 = lbase + (unsigned)(r1 * 64 + s1c * 16);

    auto issue = [&](int i) {
        unsigned go = (unsigned)(i << 6);              // i*32 halves * 2B
        unsigned lo = (unsigned)((i & 3) * 4096);
        asm volatile("global_load_async_to_lds_b128 %0, %1, %2"
                     :: "v"(l0 + lo), "v"(vbase0 + go), "s"(gB) : "memory");
        asm volatile("global_load_async_to_lds_b128 %0, %1, %2"
                     :: "v"(l1 + lo), "v"(vbase1 + go), "s"(gB) : "memory");
    };

    const _Float16* arow0 = A + (size_t)(m0 + ln) * K + hi * 16;
    const _Float16* arow1 = arow0 + (size_t)16 * K;

    auto compute = [&](int i) {
        int k0 = i << 5;
        v16h a0 = join16(*(const h8*)(arow0 + k0), *(const h8*)(arow0 + k0 + 8));
        v16h a1 = join16(*(const h8*)(arow1 + k0), *(const h8*)(arow1 + k0 + 8));
        const _Float16* bl = btile + (i & 3) * 2048 + ln * 32 + hi * 16;
        v16h bf[4];
#pragma unroll
        for (int j = 0; j < 4; ++j) bf[j] = *(const v16h*)(bl + j * 512);
#pragma unroll
        for (int j = 0; j < 4; ++j) {
            acc[0][j] = wmma_f16(a0, bf[j], acc[0][j]);
            acc[1][j] = wmma_f16(a1, bf[j], acc[1][j]);
        }
    };

    int niter = K >> 5;               // >= 32 for all our shapes
    issue(0); issue(1); issue(2);
    int steady = niter - 3;
    for (int i = 0; i < steady; ++i) {
        asm volatile("s_wait_asynccnt 4" ::: "memory");   // 6 outstanding -> oldest tile done
        __syncthreads();
        issue(i + 3);
        compute(i);
    }
    asm volatile("s_wait_asynccnt 4" ::: "memory");
    __syncthreads();
    compute(steady);
    asm volatile("s_wait_asynccnt 2" ::: "memory");
    __syncthreads();
    compute(steady + 1);
    asm volatile("s_wait_asynccnt 0" ::: "memory");
    __syncthreads();
    compute(steady + 2);

    return GemmIdx{m0, n0, hi, ln};
}

// ---------------- generic WMMA GEMM, grid = (N/64, M/128) ----------------
__global__ __launch_bounds__(128) void gemm_f16_lds_kernel(const _Float16* __restrict__ A,
                                                           const _Float16* __restrict__ Bt,
                                                           const float* __restrict__ bias,
                                                           const float* __restrict__ res,
                                                           float* __restrict__ out32,
                                                           _Float16* __restrict__ out16,
                                                           int N, int K, int flags) {
    __shared__ _Float16 btile[4][64][32];   // 4 x 4KB
    v8f acc[2][4] = {};
    GemmIdx gi = gemm_mainloop(A, Bt, K, acc, &btile[0][0][0]);
    int m0 = gi.m0, n0 = gi.n0, hi = gi.hi, ln = gi.ln;
#pragma unroll
    for (int mi = 0; mi < 2; ++mi)
#pragma unroll
    for (int j = 0; j < 4; ++j) {
        int n = n0 + j * 16 + ln;
        float bi = (flags & EPI_BIAS) ? bias[n] : 0.0f;
#pragma unroll
        for (int r = 0; r < 8; ++r) {
            int m = m0 + mi * 16 + r + hi * 8;
            float v = acc[mi][j][r] + bi;
            if (flags & EPI_RES)  v += res[(size_t)m * N + n];
            if (flags & EPI_RELU) v = fmaxf(v, 0.0f);
            if (flags & EPI_O32)  out32[(size_t)m * N + n] = v;
            if (flags & EPI_O16)  out16[(size_t)m * N + n] = (_Float16)v;
        }
    }
}

// ---------------- QKV GEMM (async-LDS): Q (scaled), K, V-transposed ----------------
// N = 3*CC, K = CC.  grid = (3*CC/64, MM/128)
__global__ __launch_bounds__(128) void gemm_qkv_kernel(const _Float16* __restrict__ A,
                                                       const _Float16* __restrict__ Bt,
                                                       _Float16* __restrict__ Qb,
                                                       _Float16* __restrict__ Kb,
                                                       _Float16* __restrict__ Vt) {
    __shared__ _Float16 btile[4][64][32];
    v8f acc[2][4] = {};
    GemmIdx gi = gemm_mainloop(A, Bt, CC, acc, &btile[0][0][0]);
    int m0 = gi.m0, n0 = gi.n0, hi = gi.hi, ln = gi.ln;
#pragma unroll
    for (int mi = 0; mi < 2; ++mi)
#pragma unroll
    for (int j = 0; j < 4; ++j) {
        int n = n0 + j * 16 + ln;
#pragma unroll
        for (int r = 0; r < 8; ++r) {
            int m = m0 + mi * 16 + r + hi * 8;
            float v = acc[mi][j][r];
            if (n < CC) {
                Qb[(size_t)m * CC + n] = (_Float16)(v * 0.125f);   // 1/sqrt(64)
            } else if (n < 2 * CC) {
                Kb[(size_t)m * CC + (n - CC)] = (_Float16)v;
            } else {
                int nv = n - 2 * CC;
                int h = nv >> 6, d = nv & 63;
                int b = m >> 11, t = m & (TT - 1);
                Vt[(size_t)((b * HH + h) * HD + d) * TT + t] = (_Float16)v;
            }
        }
    }
}

// ---------------- flash attention ----------------
// one wave per 16-row query tile; block = 4 waves; grid = (T/16/4, H, B)
__global__ __launch_bounds__(128) void attn_kernel(const _Float16* __restrict__ Qb,
                                                   const _Float16* __restrict__ Kb,
                                                   const _Float16* __restrict__ Vt,
                                                   _Float16* __restrict__ attn_h) {
    __shared__ _Float16 plds[4][16][32];
    int widx = threadIdx.x >> 5;
    int lane = threadIdx.x & 31;
    int hi = lane >> 4, ln = lane & 15;
    int qt = blockIdx.x * 4 + widx;
    int h  = blockIdx.y;
    int b  = blockIdx.z;
    int m0 = qt * 16;
    int rowbase = b * TT;

    // Q fragments (16 x 64, split over k=d in two 32-chunks)
    v16h aq[2];
#pragma unroll
    for (int dblk = 0; dblk < 2; ++dblk) {
        const _Float16* qrow = Qb + (size_t)(rowbase + m0 + ln) * CC + h * HD + dblk * 32 + hi * 16;
        aq[dblk] = join16(*(const h8*)qrow, *(const h8*)(qrow + 8));
    }

    v8f o[4] = {};
    float mrun[8], lrun[8], corr8[8];
#pragma unroll
    for (int r = 0; r < 8; ++r) { mrun[r] = NEGBIG; lrun[r] = 0.0f; }

    int nt = (m0 + 47) >> 5;   // number of 32-wide key tiles (causal)
    for (int it = 0; it < nt; ++it) {
        int s0 = it * 32;
        v8f sA = {}, sB = {};
#pragma unroll
        for (int dblk = 0; dblk < 2; ++dblk) {
            v16h bk0 = *(const v16h*)(Kb + (size_t)(rowbase + s0 + ln) * CC + h * HD + dblk * 32 + hi * 16);
            v16h bk1 = *(const v16h*)(Kb + (size_t)(rowbase + s0 + 16 + ln) * CC + h * HD + dblk * 32 + hi * 16);
            sA = wmma_f16(aq[dblk], bk0, sA);
            sB = wmma_f16(aq[dblk], bk1, sB);
        }
        bool maskit = (s0 + 31 > m0);
#pragma unroll
        for (int r = 0; r < 8; ++r) {
            int mg = m0 + r + hi * 8;
            float sv0 = sA[r], sv1 = sB[r];
            if (maskit) {
                if (s0 + ln > mg)      sv0 = NEGBIG;
                if (s0 + 16 + ln > mg) sv1 = NEGBIG;
            }
            float rm = fmaxf(sv0, sv1);
            rm = fmaxf(rm, __shfl_xor(rm, 1));
            rm = fmaxf(rm, __shfl_xor(rm, 2));
            rm = fmaxf(rm, __shfl_xor(rm, 4));
            rm = fmaxf(rm, __shfl_xor(rm, 8));
            float mnew = fmaxf(mrun[r], rm);
            float p0 = __expf(sv0 - mnew);
            float p1 = __expf(sv1 - mnew);
            float corr = __expf(mrun[r] - mnew);
            mrun[r] = mnew;
            corr8[r] = corr;
            float rs = p0 + p1;
            rs += __shfl_xor(rs, 1);
            rs += __shfl_xor(rs, 2);
            rs += __shfl_xor(rs, 4);
            rs += __shfl_xor(rs, 8);
            lrun[r] = lrun[r] * corr + rs;
            plds[widx][r + hi * 8][ln]      = (_Float16)p0;
            plds[widx][r + hi * 8][16 + ln] = (_Float16)p1;
        }
#pragma unroll
        for (int dn = 0; dn < 4; ++dn)
#pragma unroll
            for (int r = 0; r < 8; ++r) o[dn][r] *= corr8[r];

        asm volatile("s_wait_dscnt 0" ::: "memory");
        const _Float16* prow = &plds[widx][ln][hi * 16];
        v16h ap = join16(*(const h8*)prow, *(const h8*)(prow + 8));
#pragma unroll
        for (int dn = 0; dn < 4; ++dn) {
            v16h bv = *(const v16h*)(Vt + (size_t)((b * HH + h) * HD + dn * 16 + ln) * TT + s0 + hi * 16);
            o[dn] = wmma_f16(ap, bv, o[dn]);
        }
    }

    float inv[8];
#pragma unroll
    for (int r = 0; r < 8; ++r) inv[r] = 1.0f / lrun[r];
#pragma unroll
    for (int dn = 0; dn < 4; ++dn)
#pragma unroll
        for (int r = 0; r < 8; ++r)
            attn_h[(size_t)(rowbase + m0 + r + hi * 8) * CC + h * HD + dn * 16 + ln] =
                (_Float16)(o[dn][r] * inv[r]);
}

// ---------------- host launch ----------------
extern "C" void kernel_launch(void* const* d_in, const int* in_sizes, int n_in,
                              void* d_out, int out_size, void* d_ws, size_t ws_size,
                              hipStream_t stream) {
    const int*   idx     = (const int*)  d_in[0];
    const float* tok_emb = (const float*)d_in[1];
    const float* pos_emb = (const float*)d_in[2];
    const float* Wq      = (const float*)d_in[3];
    const float* Wk      = (const float*)d_in[4];
    const float* Wv      = (const float*)d_in[5];
    const float* Wo      = (const float*)d_in[6];
    const float* bo      = (const float*)d_in[7];
    const float* g1      = (const float*)d_in[8];
    const float* b1      = (const float*)d_in[9];
    const float* g2      = (const float*)d_in[10];
    const float* b2      = (const float*)d_in[11];
    const float* W_ff1   = (const float*)d_in[12];
    const float* b_ff1   = (const float*)d_in[13];
    const float* W_ff2   = (const float*)d_in[14];
    const float* b_ff2   = (const float*)d_in[15];
    const float* Wf      = (const float*)d_in[16];
    const float* bf      = (const float*)d_in[17];
    float* logits = (float*)d_out;

    // workspace carve (256B aligned)
    char* base = (char*)d_ws;
    size_t off = 0;
    auto carve = [&](size_t bytes) -> void* {
        void* p = base + off;
        off += (bytes + 255) & ~(size_t)255;
        return p;
    };
    _Float16* Wqkv_t = (_Float16*)carve((size_t)3 * CC * CC * 2);
    _Float16* Wo_t   = (_Float16*)carve((size_t)CC * CC * 2);
    _Float16* Wff1_t = (_Float16*)carve((size_t)FF * CC * 2);
    _Float16* Wff2_t = (_Float16*)carve((size_t)CC * FF * 2);
    _Float16* Wf_t   = (_Float16*)carve((size_t)VV * CC * 2);
    _Float16* xn_h   = (_Float16*)carve((size_t)MM * CC * 2);
    _Float16* Qb     = (_Float16*)carve((size_t)MM * CC * 2);
    _Float16* Kb     = (_Float16*)carve((size_t)MM * CC * 2);
    _Float16* Vt     = (_Float16*)carve((size_t)MM * CC * 2);
    _Float16* attn_h = (_Float16*)carve((size_t)MM * CC * 2);
    _Float16* xn2_h  = (_Float16*)carve((size_t)MM * CC * 2);
    _Float16* ffh    = (_Float16*)carve((size_t)MM * FF * 2);
    _Float16* x3_h   = (_Float16*)carve((size_t)MM * CC * 2);
    float*    x      = (float*)   carve((size_t)MM * CC * 4);
    float*    x2     = (float*)   carve((size_t)MM * CC * 4);
    (void)ws_size; (void)n_in; (void)in_sizes; (void)out_size;

    // 1. pack weights
    pack_qkv_kernel<<<(3 * CC * CC + 255) / 256, 256, 0, stream>>>(Wq, Wk, Wv, Wqkv_t);
    pack_t_kernel<<<(CC * CC + 255) / 256, 256, 0, stream>>>(Wo, Wo_t, CC, CC);
    pack_t_kernel<<<(FF * CC + 255) / 256, 256, 0, stream>>>(W_ff1, Wff1_t, FF, CC);
    pack_t_kernel<<<(CC * FF + 255) / 256, 256, 0, stream>>>(W_ff2, Wff2_t, CC, FF);
    pack_t_kernel<<<(VV * CC + 255) / 256, 256, 0, stream>>>(Wf, Wf_t, VV, CC);

    // 2. embed + LN1
    embed_ln_kernel<<<MM, 256, 0, stream>>>(idx, tok_emb, pos_emb, g1, b1, x, xn_h);

    // 3. QKV projection (Q pre-scaled, V stored transposed)
    gemm_qkv_kernel<<<dim3(3 * CC / 64, MM / 128), 128, 0, stream>>>(xn_h, Wqkv_t, Qb, Kb, Vt);

    // 4. flash attention
    attn_kernel<<<dim3(TT / 16 / 4, HH, BB), 128, 0, stream>>>(Qb, Kb, Vt, attn_h);

    // 5. output projection + residual -> x2 (f32)
    gemm_f16_lds_kernel<<<dim3(CC / 64, MM / 128), 128, 0, stream>>>(
        attn_h, Wo_t, bo, x, x2, nullptr, CC, CC, EPI_BIAS | EPI_RES | EPI_O32);

    // 6. LN2 -> xn2 (f16)
    ln_kernel<<<MM, 256, 0, stream>>>(x2, g2, b2, xn2_h);

    // 7. FFN1 + ReLU -> ffh (f16)
    gemm_f16_lds_kernel<<<dim3(FF / 64, MM / 128), 128, 0, stream>>>(
        xn2_h, Wff1_t, b_ff1, nullptr, nullptr, ffh, FF, CC, EPI_BIAS | EPI_RELU | EPI_O16);

    // 8. FFN2 + residual -> x3 (f16)
    gemm_f16_lds_kernel<<<dim3(CC / 64, MM / 128), 128, 0, stream>>>(
        ffh, Wff2_t, b_ff2, x2, nullptr, x3_h, CC, FF, EPI_BIAS | EPI_RES | EPI_O16);

    // 9. logits = x3 @ Wf + bf -> d_out (f32)
    gemm_f16_lds_kernel<<<dim3(VV / 64, MM / 128), 128, 0, stream>>>(
        x3_h, Wf_t, bf, nullptr, logits, nullptr, VV, CC, EPI_BIAS | EPI_O32);
}
